// SparseGroupMHA_45449343926890
// MI455X (gfx1250) — matrix operand order repaired and checked by reading, not compiled, not run on hardware
//
#include <hip/hip_runtime.h>

// ---------------- types / helpers ----------------
typedef __attribute__((ext_vector_type(16))) _Float16 v16h;
typedef __attribute__((ext_vector_type(8)))  _Float16 v8h;
typedef __attribute__((ext_vector_type(4)))  _Float16 v4h;
typedef __attribute__((ext_vector_type(8)))  float    v8f;

#define WMMA16(a, b, c) \
  __builtin_amdgcn_wmma_f32_16x16x32_f16(false, (a), false, (b), (short)0, (c), false, false)

static __device__ __forceinline__ v8f vzero8() {
  v8f z;
#pragma unroll
  for (int i = 0; i < 8; ++i) z[i] = 0.0f;
  return z;
}

// Combine two contiguous 8xf16 chunks into one 16xf16 fragment register set.
static __device__ __forceinline__ v16h ld_frag16(const _Float16* p0, const _Float16* p1) {
  v8h lo = *reinterpret_cast<const v8h*>(p0);
  v8h hi = *reinterpret_cast<const v8h*>(p1);
  return __builtin_shufflevector(lo, hi, 0, 1, 2, 3, 4, 5, 6, 7, 8, 9, 10, 11, 12, 13, 14, 15);
}

// ---------------- constants ----------------
#define TT   128
#define BB   256
#define DM   512
#define HH   8
#define DK   64
#define TOK  (TT * BB)          // 32768

// ---------------- kernel 1: X fp32 -> f16 ----------------
__global__ void __launch_bounds__(256) k_cvt_x(const float* __restrict__ in,
                                               _Float16* __restrict__ out, int n4) {
  int i = blockIdx.x * 256 + threadIdx.x;
  if (i < n4) {
    float4 f = reinterpret_cast<const float4*>(in)[i];
    v4h h = { (_Float16)f.x, (_Float16)f.y, (_Float16)f.z, (_Float16)f.w };
    reinterpret_cast<v4h*>(out)[i] = h;
  }
}

// ---------------- kernel 2: transpose all 4 weights to [N,K] f16 ----------------
__global__ void __launch_bounds__(256) k_wt(const float* __restrict__ Wq,
                                            const float* __restrict__ Wk,
                                            const float* __restrict__ Wv,
                                            const float* __restrict__ Wo,
                                            _Float16* __restrict__ out) {
  int i = blockIdx.x * 256 + threadIdx.x;   // 0 .. 4*512*512-1, = out index
  int w = i >> 18;
  int r = i & 262143;
  int n = r >> 9;
  int k = r & 511;
  const float* W = (w == 0) ? Wq : (w == 1) ? Wk : (w == 2) ? Wv : Wo;
  out[i] = (_Float16)W[k * 512 + n];        // W stored [K,N] row-major
}

// ---------------- kernel 3: QKV projection GEMM, 32x64 register tile / wave ----
// Q,K stored [t,h,b,d]; V stored transposed [t,h,d,b].  colB spans exactly one head.
__global__ void __launch_bounds__(256) k_qkv(const _Float16* __restrict__ X,
                                             const _Float16* __restrict__ WT,
                                             _Float16* __restrict__ Q,
                                             _Float16* __restrict__ K,
                                             _Float16* __restrict__ Vt) {
  const int wave = threadIdx.x >> 5;
  const int lane = threadIdx.x & 31;
  const int half = lane >> 4;
  const int l16  = lane & 15;

  int tid   = blockIdx.x * 8 + wave;   // 0 .. 3*1024*8-1
  int which = tid / 8192;              // 0=Q 1=K 2=V
  int r     = tid - which * 8192;
  int rowB  = r >> 3;                  // 0..1023 (32 rows each)
  int colB  = r & 7;                   // 0..7    (64 cols each == one head)

  const _Float16* arow0 = X + (size_t)(rowB * 32 + l16) * 512;
  const _Float16* arow1 = arow0 + 16 * 512;
  const _Float16* brow  = WT + (size_t)which * 262144 + (size_t)(colB * 64 + l16) * 512;

  v8f acc[2][4];
#pragma unroll
  for (int rt = 0; rt < 2; ++rt)
#pragma unroll
    for (int ct = 0; ct < 4; ++ct) acc[rt][ct] = vzero8();

#pragma unroll 2
  for (int k0 = 0; k0 < 512; k0 += 32) {
    v16h a0 = ld_frag16(arow0 + k0 + half * 16, arow0 + k0 + half * 16 + 8);
    v16h a1 = ld_frag16(arow1 + k0 + half * 16, arow1 + k0 + half * 16 + 8);
#pragma unroll
    for (int ct = 0; ct < 4; ++ct) {
      const _Float16* bp = brow + ct * 16 * 512 + k0 + half * 16;
      v16h b = ld_frag16(bp, bp + 8);
      acc[0][ct] = WMMA16(a0, b, acc[0][ct]);
      acc[1][ct] = WMMA16(a1, b, acc[1][ct]);
    }
  }

  // Branch-free scatter: select base pointer + offset formula arithmetically.
  const int h = colB;                              // 64 cols per head
  _Float16* dst = (which == 0) ? Q : (which == 1) ? K : Vt;
  const bool isv = (which == 2);
#pragma unroll
  for (int rt = 0; rt < 2; ++rt) {
#pragma unroll
    for (int j = 0; j < 8; ++j) {
      int grow = rowB * 32 + rt * 16 + j + 8 * half;   // global token row
      int t  = grow >> 8;
      int bb = grow & 255;
      int th = t * 8 + h;
#pragma unroll
      for (int ct = 0; ct < 4; ++ct) {
        int d = ct * 16 + l16;
        size_t off_qk = ((size_t)th * 256 + bb) * 64 + d;
        size_t off_v  = ((size_t)th * 64 + d) * 256 + bb;
        dst[isv ? off_v : off_qk] = (_Float16)acc[rt][ct][j];
      }
    }
  }
}

// ---------------- kernel 4: flash-style group-masked attention ----------------
// One wave per 16-row tile of one (t,h).  8 waves / block.
__global__ void __launch_bounds__(256) k_attn(const _Float16* __restrict__ Q,
                                              const _Float16* __restrict__ K,
                                              const _Float16* __restrict__ Vt,
                                              const int* __restrict__ gid,
                                              _Float16* __restrict__ AO) {
  __shared__ __align__(16) _Float16 pbuf[8 * 16 * 32];   // 1KB per wave

  const int wave = threadIdx.x >> 5;
  const int lane = threadIdx.x & 31;
  const int half = lane >> 4;
  const int l16  = lane & 15;

  int g   = blockIdx.x * 8 + wave;      // 0 .. 16383
  int t   = g >> 7;
  int rem = g & 127;
  int h   = rem >> 4;
  int b0  = (rem & 15) << 4;

  // Q fragments for 16 rows x 64 d, kept resident (A layout).
  const _Float16* Qrow = Q + (((size_t)(t * 8 + h) * 256) + b0 + l16) * 64;
  v16h a0 = ld_frag16(Qrow + half * 16,      Qrow + half * 16 + 8);
  v16h a1 = ld_frag16(Qrow + 32 + half * 16, Qrow + 32 + half * 16 + 8);

  // Group ids of the 8 rows this lane owns in C-fragment layout.
  int gr[8];
  {
    int4 ga = *reinterpret_cast<const int4*>(gid + b0 + 8 * half);
    int4 gb = *reinterpret_cast<const int4*>(gid + b0 + 8 * half + 4);
    gr[0] = ga.x; gr[1] = ga.y; gr[2] = ga.z; gr[3] = ga.w;
    gr[4] = gb.x; gr[5] = gb.y; gr[6] = gb.z; gr[7] = gb.w;
  }

  float m[8], l[8];
  v8f acc[4];
#pragma unroll
  for (int j = 0; j < 8; ++j) { m[j] = -1e30f; l[j] = 0.0f; }
#pragma unroll
  for (int n = 0; n < 4; ++n) acc[n] = vzero8();

  const _Float16* Kbase = K  + (size_t)(t * 8 + h) * 256 * 64;
  const _Float16* Vbase = Vt + (size_t)(t * 8 + h) * 64 * 256;
  _Float16* pb = pbuf + wave * 512;

  for (int cp = 0; cp < 256; cp += 32) {
    // ---- scores for two 16-col tiles: S = Q K^T ----
    v8f s0, s1;
    {
      const _Float16* kr = Kbase + (size_t)(cp + l16) * 64;
      v16h bk0 = ld_frag16(kr + half * 16, kr + half * 16 + 8);
      v16h bk1 = ld_frag16(kr + 32 + half * 16, kr + 32 + half * 16 + 8);
      s0 = WMMA16(a0, bk0, vzero8());
      s0 = WMMA16(a1, bk1, s0);
    }
    {
      const _Float16* kr = Kbase + (size_t)(cp + 16 + l16) * 64;
      v16h bk0 = ld_frag16(kr + half * 16, kr + half * 16 + 8);
      v16h bk1 = ld_frag16(kr + 32 + half * 16, kr + 32 + half * 16 + 8);
      s1 = WMMA16(a0, bk0, vzero8());
      s1 = WMMA16(a1, bk1, s1);
    }

    // ---- scale + group mask (large-negative sentinel; self-corrects, see notes) ----
    int gc0 = gid[cp + l16];
    int gc1 = gid[cp + 16 + l16];
#pragma unroll
    for (int j = 0; j < 8; ++j) {
      s0[j] = (gr[j] == gc0) ? s0[j] * 0.125f : -1e30f;
      s1[j] = (gr[j] == gc1) ? s1[j] * 0.125f : -1e30f;
    }

    // ---- online softmax update (row reductions across 16 lanes of the half-wave) ----
    float p0[8], p1[8];
#pragma unroll
    for (int j = 0; j < 8; ++j) {
      float v = fmaxf(s0[j], s1[j]);
      v = fmaxf(v, __shfl_xor(v, 1, 16));
      v = fmaxf(v, __shfl_xor(v, 2, 16));
      v = fmaxf(v, __shfl_xor(v, 4, 16));
      v = fmaxf(v, __shfl_xor(v, 8, 16));
      float nm = fmaxf(m[j], v);
      float sc = __expf(m[j] - nm);
      m[j] = nm;
      float e0 = __expf(s0[j] - nm);
      float e1 = __expf(s1[j] - nm);
      p0[j] = e0; p1[j] = e1;
      float rs = e0 + e1;
      rs += __shfl_xor(rs, 1, 16);
      rs += __shfl_xor(rs, 2, 16);
      rs += __shfl_xor(rs, 4, 16);
      rs += __shfl_xor(rs, 8, 16);
      l[j] = l[j] * sc + rs;
      acc[0][j] *= sc; acc[1][j] *= sc; acc[2][j] *= sc; acc[3][j] *= sc;
    }

    // ---- C-layout -> A-layout for P via per-wave LDS bounce ----
#pragma unroll
    for (int j = 0; j < 8; ++j) {
      int rowoff = (j + 8 * half) * 32;
      pb[rowoff + l16]      = (_Float16)p0[j];
      pb[rowoff + 16 + l16] = (_Float16)p1[j];
    }
    __asm__ volatile("s_wait_dscnt 0" ::: "memory");  // same-wave DS ordering + compiler fence
    v16h ap = ld_frag16(pb + l16 * 32 + half * 16, pb + l16 * 32 + half * 16 + 8);
    __asm__ volatile("" ::: "memory");

    // ---- O += P * V  (V stored [d, b] so K-dim is contiguous) ----
#pragma unroll
    for (int n = 0; n < 4; ++n) {
      const _Float16* vr = Vbase + (size_t)(n * 16 + l16) * 256 + cp;
      v16h bv = ld_frag16(vr + half * 16, vr + half * 16 + 8);
      acc[n] = WMMA16(ap, bv, acc[n]);
    }
  }

  // ---- epilogue: divide by row sums, store [token, h*64+d] f16 ----
#pragma unroll
  for (int j = 0; j < 8; ++j) {
    float rl = 1.0f / l[j];
    int token = t * 256 + b0 + j + 8 * half;
#pragma unroll
    for (int n = 0; n < 4; ++n) {
      AO[(size_t)token * 512 + h * 64 + n * 16 + l16] = (_Float16)(acc[n][j] * rl);
    }
  }
}

// ---------------- kernel 5: output projection GEMM, 32x64 tile / wave (f32 out) --
__global__ void __launch_bounds__(256) k_out(const _Float16* __restrict__ AO,
                                             const _Float16* __restrict__ WoT,
                                             float* __restrict__ out) {
  const int wave = threadIdx.x >> 5;
  const int lane = threadIdx.x & 31;
  const int half = lane >> 4;
  const int l16  = lane & 15;

  int tid  = blockIdx.x * 8 + wave;    // 0 .. 8191
  int rowB = tid >> 3;                 // 0..1023
  int colB = tid & 7;                  // 0..7

  const _Float16* arow0 = AO + (size_t)(rowB * 32 + l16) * 512;
  const _Float16* arow1 = arow0 + 16 * 512;
  const _Float16* brow  = WoT + (size_t)(colB * 64 + l16) * 512;

  v8f acc[2][4];
#pragma unroll
  for (int rt = 0; rt < 2; ++rt)
#pragma unroll
    for (int ct = 0; ct < 4; ++ct) acc[rt][ct] = vzero8();

#pragma unroll 2
  for (int k0 = 0; k0 < 512; k0 += 32) {
    v16h a0 = ld_frag16(arow0 + k0 + half * 16, arow0 + k0 + half * 16 + 8);
    v16h a1 = ld_frag16(arow1 + k0 + half * 16, arow1 + k0 + half * 16 + 8);
#pragma unroll
    for (int ct = 0; ct < 4; ++ct) {
      const _Float16* bp = brow + ct * 16 * 512 + k0 + half * 16;
      v16h b = ld_frag16(bp, bp + 8);
      acc[0][ct] = WMMA16(a0, b, acc[0][ct]);
      acc[1][ct] = WMMA16(a1, b, acc[1][ct]);
    }
  }

#pragma unroll
  for (int rt = 0; rt < 2; ++rt) {
#pragma unroll
    for (int j = 0; j < 8; ++j) {
      int grow = rowB * 32 + rt * 16 + j + 8 * half;
#pragma unroll
      for (int ct = 0; ct < 4; ++ct) {
        out[(size_t)grow * 512 + colB * 64 + ct * 16 + l16] = acc[rt][ct][j];
      }
    }
  }
}

// ---------------- launch ----------------
extern "C" void kernel_launch(void* const* d_in, const int* in_sizes, int n_in,
                              void* d_out, int out_size, void* d_ws, size_t ws_size,
                              hipStream_t stream) {
  const float* X   = (const float*)d_in[0];
  // d_in[1] = mask (unused)
  const int*   gid = (const int*)d_in[2];
  const float* Wq  = (const float*)d_in[3];
  const float* Wk  = (const float*)d_in[4];
  const float* Wv  = (const float*)d_in[5];
  const float* Wo  = (const float*)d_in[6];
  float* out = (float*)d_out;

  char* ws = (char*)d_ws;
  const size_t MB = 1024ull * 1024ull;
  _Float16* Xh  = (_Float16*)(ws);                 // 32 MB  [TOK, 512]
  _Float16* WT  = (_Float16*)(ws + 32 * MB);       //  2 MB  4 x [512, 512] (N-major)
  _Float16* Qh  = (_Float16*)(ws + 34 * MB);       // 32 MB  [t,h,b,d]
  _Float16* Kh  = (_Float16*)(ws + 66 * MB);       // 32 MB  [t,h,b,d]
  _Float16* Vth = (_Float16*)(ws + 98 * MB);       // 32 MB  [t,h,d,b]
  _Float16* AO  = (_Float16*)(ws + 130 * MB);      // 32 MB  [TOK, 512]
  // total 162 MB

  k_cvt_x<<<(TOK * DM / 4 + 255) / 256, 256, 0, stream>>>(X, Xh, TOK * DM / 4);
  k_wt   <<<(4 * 512 * 512) / 256, 256, 0, stream>>>(Wq, Wk, Wv, Wo, WT);
  k_qkv  <<<3 * 1024 * 8 / 8, 256, 0, stream>>>(Xh, WT, Qh, Kh, Vth);
  k_attn <<<16384 / 8, 256, 0, stream>>>(Qh, Kh, Vth, gid, AO);
  k_out  <<<1024, 256, 0, stream>>>(AO, WT + 3 * 262144, out);
}
